// LowResNet_80152679678556
// MI455X (gfx1250) — matrix operand order repaired and checked by reading, not compiled, not run on hardware
//
#include <hip/hip_runtime.h>
#include <cstdint>
#include <cstddef>

typedef __attribute__((ext_vector_type(16))) __bf16 v16bf;
typedef __attribute__((ext_vector_type(8)))  float  v8f;

#define C_DIM 512
#define H_DIM 256
#define B_DIM 64
#define S_DIM 12
#define N_DIM 4096
#define M_GRU 32768           // B_DIM * C_DIM
#define AS_STRIDE 264         // 256 + 8 pad floats

// ---------------------------------------------------------------------------
// Generic WMMA GEMM:  Y[m][o] = bias[o] + sum_k A[m][k]*Bp[k][o]
//                              (+ sum_k A2[m][k]*Bp2[k][o])
//                              (+ xp0[m]*wx0[o] + xp1[m]*wx1[o])
// A: f32 row-major (converted to bf16 on the fly). Bp: bf16 pre-packed in
// per-lane WMMA fragment order. Epilogue selected by mode.
// ---------------------------------------------------------------------------
struct GArgs {
  const float*  A;   int lda;
  const float*  A2;  int lda2;
  const __bf16* Bp;  const __bf16* Bp2;
  const float*  bias;
  const float*  xp0; int xp0_sB; const float* wx0; int wx0_ld;
  const float*  xp1; int xp1_sB; const float* wx1; int wx1_ld;
  float* Yout; int ldo;
  const float* hbuf; float* hout; float* rhbuf; float* ubuf;
  int M, K, O, mode;   // mode 0: store, 1: gate, 2: cand, 3: out-scatter
};

// A fragment (16x32 bf16), gfx1250 layout:
// lanes 0-15: K = k0+{0..7, 16..23}; lanes 16-31: K = k0+{8..15, 24..31}
__device__ __forceinline__ v16bf load_a_frag(const float* __restrict__ As,
                                             int arow, int ks, int kk) {
  const float* ap = &As[arow * AS_STRIDE + (ks << 5) + kk];
  float4 a0 = *(const float4*)(ap);
  float4 a1 = *(const float4*)(ap + 4);
  float4 a2 = *(const float4*)(ap + 16);
  float4 a3 = *(const float4*)(ap + 20);
  v16bf af;
  af[0]=(__bf16)a0.x;  af[1]=(__bf16)a0.y;  af[2]=(__bf16)a0.z;  af[3]=(__bf16)a0.w;
  af[4]=(__bf16)a1.x;  af[5]=(__bf16)a1.y;  af[6]=(__bf16)a1.z;  af[7]=(__bf16)a1.w;
  af[8]=(__bf16)a2.x;  af[9]=(__bf16)a2.y;  af[10]=(__bf16)a2.z; af[11]=(__bf16)a2.w;
  af[12]=(__bf16)a3.x; af[13]=(__bf16)a3.y; af[14]=(__bf16)a3.z; af[15]=(__bf16)a3.w;
  return af;
}

__device__ __forceinline__ void load_b_frags(const __bf16* __restrict__ Bop,
                                             int ktile, int ONT, int o16, int lane,
                                             v16bf* __restrict__ b) {
  const __bf16* bb = Bop + (((size_t)ktile * ONT + o16) << 9) + (lane << 4);
  #pragma unroll
  for (int nt = 0; nt < 4; ++nt) b[nt] = *(const v16bf*)(bb + ((size_t)nt << 9));
}

__global__ __launch_bounds__(128) void gemm_wmma_k(GArgs g) {
  __shared__ float As[64 * AS_STRIDE];
  const int tid  = threadIdx.x;
  const int wid  = tid >> 5;
  const int lane = tid & 31;
  const int m0   = blockIdx.y << 6;
  const int o0   = blockIdx.x << 6;
  const int o16  = o0 >> 4;
  const int ONT  = g.O >> 4;

  v8f acc[4];
  #pragma unroll
  for (int i = 0; i < 4; ++i)
    #pragma unroll
    for (int j = 0; j < 8; ++j) acc[i][j] = 0.0f;

  const int arow = (wid << 4) + (lane & 15);
  const int kk   = (lane < 16) ? 0 : 8;

  for (int src = 0; src < 2; ++src) {
    const float*  Aop = src ? g.A2  : g.A;
    const __bf16* Bop = src ? g.Bp2 : g.Bp;
    const int     lda = src ? g.lda2 : g.lda;
    if (Aop == nullptr) continue;          // uniform (kernarg) branch
    for (int kc = 0; kc < g.K; kc += 256) {
      __syncthreads();
      {   // cooperative stage: 64 rows x 256 cols f32 -> LDS
        const float* gbase = Aop + (size_t)m0 * lda + kc;
        #pragma unroll
        for (int i = 0; i < 32; ++i) {
          int idx = tid + (i << 7);
          int r   = idx >> 6;
          int c4  = (idx & 63) << 2;
          const float4 v = *(const float4*)(gbase + (size_t)r * lda + c4);
          *(float4*)&As[r * AS_STRIDE + c4] = v;
        }
      }
      __syncthreads();
      const int ktb = kc >> 5;
      // software-pipelined: prefetch fragments of step ks+1 while WMMAs of
      // step ks execute, so loads never drain to loadcnt==0 at the WMMA.
      v16bf acur = load_a_frag(As, arow, 0, kk);
      v16bf bcur[4];
      load_b_frags(Bop, ktb, ONT, o16, lane, bcur);
      #pragma unroll
      for (int ks = 0; ks < 7; ++ks) {
        v16bf anxt = load_a_frag(As, arow, ks + 1, kk);
        v16bf bnxt[4];
        load_b_frags(Bop, ktb + ks + 1, ONT, o16, lane, bnxt);
        #pragma unroll
        for (int nt = 0; nt < 4; ++nt)
          acc[nt] = __builtin_amdgcn_wmma_f32_16x16x32_bf16(
              false, acur, false, bcur[nt], (short)0, acc[nt], false, false);
        acur = anxt;
        #pragma unroll
        for (int nt = 0; nt < 4; ++nt) bcur[nt] = bnxt[nt];
      }
      #pragma unroll
      for (int nt = 0; nt < 4; ++nt)      // tail step (ks == 7)
        acc[nt] = __builtin_amdgcn_wmma_f32_16x16x32_bf16(
            false, acur, false, bcur[nt], (short)0, acc[nt], false, false);
    }
  }

  // Epilogue. C/D layout: VGPR v, lane L -> row = (L<16 ? v : v+8), col = L%16
  const int col   = lane & 15;
  const int rb    = (lane < 16) ? 0 : 8;
  const int mbase = m0 + (wid << 4);
  #pragma unroll
  for (int nt = 0; nt < 4; ++nt) {
    const int o = o0 + (nt << 4) + col;
    float bsv = g.bias ? g.bias[o] : 0.0f;
    float w0  = g.wx0 ? g.wx0[(size_t)o * g.wx0_ld] : 0.0f;
    float w1  = g.wx1 ? g.wx1[(size_t)o * g.wx1_ld] : 0.0f;
    #pragma unroll
    for (int v = 0; v < 8; ++v) {
      const int m = mbase + rb + v;
      float y = acc[nt][v] + bsv;
      if (g.xp0) y += g.xp0[(size_t)(m >> 9) * g.xp0_sB + (m & 511)] * w0;
      if (g.xp1) y += g.xp1[(size_t)(m >> 9) * g.xp1_sB + (m & 511)] * w1;
      if (g.mode == 0) {
        g.Yout[(size_t)m * g.ldo + o] = y;
      } else if (g.mode == 1) {               // gate: r (o<H) -> r*h ; u (o>=H)
        float sg = 1.0f / (1.0f + __expf(-y));
        if (o < H_DIM) g.rhbuf[(size_t)m * H_DIM + o] = sg * g.hbuf[(size_t)m * H_DIM + o];
        else           g.ubuf[(size_t)m * H_DIM + (o - H_DIM)] = sg;
      } else if (g.mode == 2) {               // candidate: h = u*h + (1-u)*tanh(y)
        float cd = tanhf(y);
        size_t ix = (size_t)m * H_DIM + o;
        float uu = g.ubuf[ix];
        g.hout[ix] = uu * g.hbuf[ix] + (1.0f - uu) * cd;
      } else {                                // out[b,0,n,s], m=(b*12+s), o=n
        g.Yout[(size_t)(m / S_DIM) * (N_DIM * S_DIM) + (size_t)o * S_DIM + (m % S_DIM)] = y;
      }
    }
  }
}

// ---------------------------------------------------------------------------
// Prep kernels
// ---------------------------------------------------------------------------
__global__ void zero_f32_k(float* p, int n) {
  int i = blockIdx.x * blockDim.x + threadIdx.x;
  if (i < n) p[i] = 0.0f;
}

__global__ void rowsum_abs_k(const float* __restrict__ I, float* __restrict__ sums) {
  __shared__ float red[256];
  int c = blockIdx.x;
  float s = 0.0f;
  for (int n = threadIdx.x; n < N_DIM; n += 256) s += fabsf(I[(size_t)c * N_DIM + n]);
  red[threadIdx.x] = s;
  __syncthreads();
  for (int st = 128; st > 0; st >>= 1) {
    if (threadIdx.x < st) red[threadIdx.x] += red[threadIdx.x + st];
    __syncthreads();
  }
  if (threadIdx.x == 0) sums[c] = red[0];
}

// Pack B[k][o] into per-lane WMMA fragment order:
//   idx = ((kt*ONT + nt)*32 + lane)*16 + e ; k = kt*32 + (lane<16?0:8) + (e<8?e:e+8) ; o = nt*16 + lane%16
__global__ void pack_b_k(const float* __restrict__ W, int ldw, int col0, int K, int O,
                         int trans, const float* __restrict__ rowscale,
                         __bf16* __restrict__ out) {
  int idx = blockIdx.x * 256 + threadIdx.x;
  if (idx >= K * O) return;
  int e    = idx & 15;
  int rest = idx >> 4;
  int lane = rest & 31; rest >>= 5;
  int ONT  = O >> 4;
  int nt   = rest % ONT;
  int kt   = rest / ONT;
  int k    = (kt << 5) + ((lane < 16) ? 0 : 8) + ((e < 8) ? e : e + 8);
  int o    = (nt << 4) + (lane & 15);
  float v = trans ? W[(size_t)(col0 + k) * ldw + o] : W[(size_t)o * ldw + col0 + k];
  if (rowscale) v /= rowscale[o];
  out[idx] = (__bf16)v;
}

// x (B,F,N,S) -> Xr[m=(b*F+f)*S+s][n]
__global__ void relayout_x_k(const float* __restrict__ x, float* __restrict__ Xr) {
  int idx = blockIdx.x * 256 + threadIdx.x;
  if (idx >= 1536 * N_DIM) return;
  int n  = idx & (N_DIM - 1);
  int m  = idx >> 12;
  int s  = m % S_DIM;
  int bf = m / S_DIM;
  Xr[idx] = x[((size_t)bf * N_DIM + n) * S_DIM + s];
}

// y[b,0,c] = pb + dot(projW, h1[m=(b,c)][:]) ; store Ysb[(b*12+t)*512 + c]
__global__ void proj_k(const float* __restrict__ h1, const float* __restrict__ pw,
                       const float* __restrict__ pb, float* __restrict__ Ysb, int t) {
  int m = blockIdx.x * blockDim.x + threadIdx.x;
  if (m >= M_GRU) return;
  const float* hp = h1 + (size_t)m * H_DIM;
  float s = pb[0];
  for (int i = 0; i < H_DIM; i += 4) {
    float4 h4 = *(const float4*)(hp + i);
    float4 w4 = *(const float4*)(pw + i);
    s += h4.x * w4.x + h4.y * w4.y + h4.z * w4.z + h4.w * w4.w;
  }
  int b = m >> 9, c = m & 511;
  Ysb[((size_t)b * S_DIM + t) * C_DIM + c] = s;
}

// ---------------------------------------------------------------------------
extern "C" void kernel_launch(void* const* d_in, const int* in_sizes, int n_in,
                              void* d_out, int out_size, void* d_ws, size_t ws_size,
                              hipStream_t stream) {
  if (n_in < 20) return;
  const float* x = (const float*)d_in[0];
  const float* I = (const float*)d_in[1];

  struct LayerP { const float* Wg; const float* bg; const float* Wc; const float* bc; };
  LayerP enc[2] = {}, dec[2] = {};
  bool firstIsEnc = false;
  for (int i = 2; i < 10; ++i)
    if (in_sizes[i] == 66048 || in_sizes[i] == 132096) firstIsEnc = true;
  auto fill = [&](int base, LayerP& L) {
    for (int j = 0; j < 4; ++j) {
      int sz = in_sizes[base + j];
      const float* p = (const float*)d_in[base + j];
      if (sz == 512) L.bg = p;
      else if (sz == 256) L.bc = p;
      else if (sz == 66048 || sz == 65792 || sz == 131072) L.Wc = p;
      else L.Wg = p;
    }
  };
  const int encBase = firstIsEnc ? 2 : 10;
  const int decBase = firstIsEnc ? 10 : 2;
  fill(encBase, enc[0]); fill(encBase + 4, enc[1]);
  fill(decBase, dec[0]); fill(decBase + 4, dec[1]);
  const float* projW = (const float*)d_in[18];
  const float* projb = (const float*)d_in[19];
  const int ld_e0 = 258, ld_d0 = 257, ld_l1 = 512;

  // ---- workspace bump allocator ----
  char* w = (char*)d_ws;
  auto bump = [&](size_t bytes) { char* r = w; w += (bytes + 255) & ~(size_t)255; return r; };
  float*  sums  = (float*) bump(512 * 4);
  __bf16* DIp   = (__bf16*)bump((size_t)N_DIM * 512 * 2);
  __bf16* Ip    = (__bf16*)bump((size_t)512 * N_DIM * 2);
  __bf16* e0WgH = (__bf16*)bump((size_t)256 * 512 * 2);
  __bf16* e0WcH = (__bf16*)bump((size_t)256 * 256 * 2);
  __bf16* e1WgX = (__bf16*)bump((size_t)256 * 512 * 2);
  __bf16* e1WgH = (__bf16*)bump((size_t)256 * 512 * 2);
  __bf16* e1WcX = (__bf16*)bump((size_t)256 * 256 * 2);
  __bf16* e1WcH = (__bf16*)bump((size_t)256 * 256 * 2);
  __bf16* d0WgH = (__bf16*)bump((size_t)256 * 512 * 2);
  __bf16* d0WcH = (__bf16*)bump((size_t)256 * 256 * 2);
  __bf16* d1WgX = (__bf16*)bump((size_t)256 * 512 * 2);
  __bf16* d1WgH = (__bf16*)bump((size_t)256 * 512 * 2);
  __bf16* d1WcX = (__bf16*)bump((size_t)256 * 256 * 2);
  __bf16* d1WcH = (__bf16*)bump((size_t)256 * 256 * 2);
  float* Xr  = (float*)bump((size_t)1536 * N_DIM * 4);
  float* Xc  = (float*)bump((size_t)1536 * C_DIM * 4);
  float* h0  = (float*)bump((size_t)M_GRU * H_DIM * 4);
  float* h1  = (float*)bump((size_t)M_GRU * H_DIM * 4);
  float* ub  = (float*)bump((size_t)M_GRU * H_DIM * 4);
  float* rh  = (float*)bump((size_t)M_GRU * H_DIM * 4);
  float* Ysb = (float*)bump((size_t)768 * C_DIM * 4);
  (void)ws_size; (void)out_size;

  // ---- prep ----
  zero_f32_k<<<32768, 256, 0, stream>>>(h0, M_GRU * H_DIM);
  zero_f32_k<<<32768, 256, 0, stream>>>(h1, M_GRU * H_DIM);
  rowsum_abs_k<<<512, 256, 0, stream>>>(I, sums);
  auto pack = [&](const float* W, int ldw, int col0, int K, int O, int trans,
                  const float* rs, __bf16* out) {
    int total = K * O;
    pack_b_k<<<(total + 255) / 256, 256, 0, stream>>>(W, ldw, col0, K, O, trans, rs, out);
  };
  pack(I, N_DIM, 0, N_DIM, 512, 0, sums, DIp);      // DI^T packed (k=n, o=c)
  pack(I, N_DIM, 0, 512, N_DIM, 1, nullptr, Ip);    // I packed    (k=c, o=n)
  pack(enc[0].Wg, ld_e0, 2,   256, 512, 0, nullptr, e0WgH);
  pack(enc[0].Wc, ld_e0, 2,   256, 256, 0, nullptr, e0WcH);
  pack(enc[1].Wg, ld_l1, 0,   256, 512, 0, nullptr, e1WgX);
  pack(enc[1].Wg, ld_l1, 256, 256, 512, 0, nullptr, e1WgH);
  pack(enc[1].Wc, ld_l1, 0,   256, 256, 0, nullptr, e1WcX);
  pack(enc[1].Wc, ld_l1, 256, 256, 256, 0, nullptr, e1WcH);
  pack(dec[0].Wg, ld_d0, 1,   256, 512, 0, nullptr, d0WgH);
  pack(dec[0].Wc, ld_d0, 1,   256, 256, 0, nullptr, d0WcH);
  pack(dec[1].Wg, ld_l1, 0,   256, 512, 0, nullptr, d1WgX);
  pack(dec[1].Wg, ld_l1, 256, 256, 512, 0, nullptr, d1WgH);
  pack(dec[1].Wc, ld_l1, 0,   256, 256, 0, nullptr, d1WcX);
  pack(dec[1].Wc, ld_l1, 256, 256, 256, 0, nullptr, d1WcH);
  relayout_x_k<<<(1536 * N_DIM + 255) / 256, 256, 0, stream>>>(x, Xr);

  auto gemm = [&](GArgs g) {
    dim3 grid(g.O / 64, g.M / 64);
    gemm_wmma_k<<<grid, 128, 0, stream>>>(g);
  };

  // ---- front projection: Xc[m=(b*2+f)*12+s][c] ----
  { GArgs g = {}; g.A = Xr; g.lda = N_DIM; g.Bp = DIp; g.Yout = Xc; g.ldo = C_DIM;
    g.M = 1536; g.K = N_DIM; g.O = 512; g.mode = 0; gemm(g); }

  // ---- encoder: 12 steps x 2 layers ----
  for (int s = 0; s < S_DIM; ++s) {
    { GArgs g = {}; g.A = h0; g.lda = H_DIM; g.Bp = e0WgH; g.bias = enc[0].bg;
      g.xp0 = Xc + (size_t)s * 512;        g.xp0_sB = 12288; g.wx0 = enc[0].Wg;     g.wx0_ld = ld_e0;
      g.xp1 = Xc + (size_t)(12 + s) * 512; g.xp1_sB = 12288; g.wx1 = enc[0].Wg + 1; g.wx1_ld = ld_e0;
      g.hbuf = h0; g.rhbuf = rh; g.ubuf = ub;
      g.M = M_GRU; g.K = 256; g.O = 512; g.mode = 1; gemm(g); }
    { GArgs g = {}; g.A = rh; g.lda = H_DIM; g.Bp = e0WcH; g.bias = enc[0].bc;
      g.xp0 = Xc + (size_t)s * 512;        g.xp0_sB = 12288; g.wx0 = enc[0].Wc;     g.wx0_ld = ld_e0;
      g.xp1 = Xc + (size_t)(12 + s) * 512; g.xp1_sB = 12288; g.wx1 = enc[0].Wc + 1; g.wx1_ld = ld_e0;
      g.hbuf = h0; g.hout = h0; g.ubuf = ub;
      g.M = M_GRU; g.K = 256; g.O = 256; g.mode = 2; gemm(g); }
    { GArgs g = {}; g.A = h1; g.lda = H_DIM; g.Bp = e1WgH;
      g.A2 = h0; g.lda2 = H_DIM; g.Bp2 = e1WgX; g.bias = enc[1].bg;
      g.hbuf = h1; g.rhbuf = rh; g.ubuf = ub;
      g.M = M_GRU; g.K = 256; g.O = 512; g.mode = 1; gemm(g); }
    { GArgs g = {}; g.A = rh; g.lda = H_DIM; g.Bp = e1WcH;
      g.A2 = h0; g.lda2 = H_DIM; g.Bp2 = e1WcX; g.bias = enc[1].bc;
      g.hbuf = h1; g.hout = h1; g.ubuf = ub;
      g.M = M_GRU; g.K = 256; g.O = 256; g.mode = 2; gemm(g); }
  }

  // ---- decoder: 12 steps ----
  for (int t = 0; t < S_DIM; ++t) {
    const float* yb; int ysB;
    if (t == 0) { yb = Xc + (size_t)11 * 512; ysB = 12288; }        // y0 = xc[:,0,:,-1]
    else        { yb = Ysb + (size_t)(t - 1) * 512; ysB = 6144; }   // previous projection
    { GArgs g = {}; g.A = h0; g.lda = H_DIM; g.Bp = d0WgH; g.bias = dec[0].bg;
      g.xp0 = yb; g.xp0_sB = ysB; g.wx0 = dec[0].Wg; g.wx0_ld = ld_d0;
      g.hbuf = h0; g.rhbuf = rh; g.ubuf = ub;
      g.M = M_GRU; g.K = 256; g.O = 512; g.mode = 1; gemm(g); }
    { GArgs g = {}; g.A = rh; g.lda = H_DIM; g.Bp = d0WcH; g.bias = dec[0].bc;
      g.xp0 = yb; g.xp0_sB = ysB; g.wx0 = dec[0].Wc; g.wx0_ld = ld_d0;
      g.hbuf = h0; g.hout = h0; g.ubuf = ub;
      g.M = M_GRU; g.K = 256; g.O = 256; g.mode = 2; gemm(g); }
    { GArgs g = {}; g.A = h1; g.lda = H_DIM; g.Bp = d1WgH;
      g.A2 = h0; g.lda2 = H_DIM; g.Bp2 = d1WgX; g.bias = dec[1].bg;
      g.hbuf = h1; g.rhbuf = rh; g.ubuf = ub;
      g.M = M_GRU; g.K = 256; g.O = 512; g.mode = 1; gemm(g); }
    { GArgs g = {}; g.A = rh; g.lda = H_DIM; g.Bp = d1WcH;
      g.A2 = h0; g.lda2 = H_DIM; g.Bp2 = d1WcX; g.bias = dec[1].bc;
      g.hbuf = h1; g.hout = h1; g.ubuf = ub;
      g.M = M_GRU; g.K = 256; g.O = 256; g.mode = 2; gemm(g); }
    proj_k<<<128, 256, 0, stream>>>(h1, projW, projb, Ysb, t);
  }

  // ---- back projection: out[b,0,n,s] = sum_c Ysb[(b,s)][c] * I[c][n] ----
  { GArgs g = {}; g.A = Ysb; g.lda = C_DIM; g.Bp = Ip; g.Yout = (float*)d_out;
    g.M = 768; g.K = 512; g.O = N_DIM; g.mode = 3; gemm(g); }
}
